// GRUNetwork_58600533786907
// MI455X (gfx1250) — matrix operand order repaired and checked by reading, not compiled, not run on hardware
//
#include <hip/hip_runtime.h>
#include <math.h>

// ---------------- problem constants (from reference) ----------------
constexpr int Bc   = 512;
constexpr int Gc   = 10;
constexpr int Tc   = 100;
constexpr int Dc   = 50;   // input feature dim
constexpr int Hc   = 30;   // hidden
constexpr int H3   = 90;   // 3*H
constexpr int EMB  = 20;
constexpr int NEMB = 10;
constexpr int OUT  = 10;
constexpr int HALFSEQ = Bc * Gc;        // 5120 sequences per input tensor
constexpr int NSEQ    = 2 * HALFSEQ;    // 10240
constexpr int NGRP    = NSEQ / 16;      // 640 groups of 16 sequences

// ---------------- WMMA vector types (gfx1250, wave32) ----------------
typedef __attribute__((ext_vector_type(16))) __bf16 v16bf;
typedef __attribute__((ext_vector_type(8)))  __bf16 v8bf;
typedef __attribute__((ext_vector_type(8)))  float  v8f;

__device__ __forceinline__ unsigned short f2bf_bits(float f) {
    union { float f; unsigned u; } x; x.f = f;
    unsigned r = x.u + 0x7FFFu + ((x.u >> 16) & 1u);     // RNE
    return (unsigned short)(r >> 16);
}
__device__ __forceinline__ __bf16 f2bf(float f) {
    unsigned short h = f2bf_bits(f);
    return __builtin_bit_cast(__bf16, h);
}
// pack two fp32 -> two bf16 in one dword (for a single ds_store_b32)
__device__ __forceinline__ unsigned f2bf_pack(float a, float b) {
    return (unsigned)f2bf_bits(a) | ((unsigned)f2bf_bits(b) << 16);
}

__device__ __forceinline__ v16bf cat8(v8bf lo, v8bf hi) {
    return __builtin_shufflevector(lo, hi, 0,1,2,3,4,5,6,7,8,9,10,11,12,13,14,15);
}

// Fast gate nonlinearities: v_exp_f32 + v_rcp_f32 instead of libm expf and
// the IEEE v_div_scale/v_div_fmas divide chain (these run 100x per group on
// the sequential critical path).
__device__ __forceinline__ float sigmoid_fast(float s) {
    // exp overflow -> inf -> rcp(inf) = 0: still correct, no NaN.
    return __builtin_amdgcn_rcpf(1.0f + __expf(-s));
}
__device__ __forceinline__ float tanh_fast(float x) {
    x = fminf(fmaxf(x, -15.0f), 15.0f);          // keep (1-e)*rcp(1+e) NaN-free
    float e = __expf(-2.0f * x);
    return (1.0f - e) * __builtin_amdgcn_rcpf(1.0f + e);
}

// =====================================================================
// Kernel 1: fused GRU over all 10240 sequences.
// Block = 192 threads (6 waves). Each block owns 16 sequences; wave w owns
// 16-column tile w of the 96-wide (padded 3H) gate dimension.
// Matmuls: v_wmma_f32_16x16x32_bf16 (bf16 inputs, fp32 accumulate).
// =====================================================================
__global__ __launch_bounds__(192)
void gru_kernel(const float* __restrict__ in1, const float* __restrict__ in2,
                const float* __restrict__ Wih, const float* __restrict__ Whh,
                const float* __restrict__ bih, const float* __restrict__ bhh,
                float* __restrict__ emb1_out,   // d_out + 5120 : [5120*30]
                float* __restrict__ emb2_ws)    // workspace    : [5120*30]
{
    __shared__ __bf16 sWih[96][64];   // W_ih padded (rows=out col, cols=K)  = B frags
    __shared__ __bf16 sWhh[96][32];   // W_hh padded                          = B frags
    __shared__ __bf16 sX[16][64];     // x_t tile, bf16, K padded 50->64      = A frags
    __shared__ __bf16 sH[16][32];     // h tile, bf16, K padded 30->32        = A frags
    __shared__ float  sSumX[16][96];  // x_t @ W_ih^T
    __shared__ float  sSumH[16][96];  // h   @ W_hh^T
    __shared__ float  sHf[16][32];    // fp32 hidden state
    __shared__ float  sBih[96], sBhh[96];

    const int tid  = threadIdx.x;
    const int lane = tid & 31;
    const int wv   = tid >> 5;                 // 0..5 -> column tile
    const int s0   = blockIdx.x * 16;          // first global sequence of group

    const float* xb;
    int soff;
    if (s0 < HALFSEQ) { xb = in1; soff = s0; }
    else              { xb = in2; soff = s0 - HALFSEQ; }

    // ---- stage weights/biases in LDS (bf16, zero padded) ----
    for (int e = tid; e < 96 * 64; e += 192) {
        int r = e >> 6, k = e & 63;
        float v = (r < H3 && k < Dc) ? Wih[r * Dc + k] : 0.0f;
        sWih[r][k] = f2bf(v);
    }
    for (int e = tid; e < 96 * 32; e += 192) {
        int r = e >> 5, k = e & 31;
        float v = (r < H3 && k < Hc) ? Whh[r * Hc + k] : 0.0f;
        sWhh[r][k] = f2bf(v);
    }
    for (int e = tid; e < 96; e += 192) {
        sBih[e] = (e < H3) ? bih[e] : 0.0f;
        sBhh[e] = (e < H3) ? bhh[e] : 0.0f;
    }
    for (int e = tid; e < 16 * 32; e += 192) {
        int i = e >> 5, k = e & 31;
        sHf[i][k] = 0.0f;
        sH[i][k]  = f2bf(0.0f);
    }
    for (int e = tid; e < 16 * 64; e += 192) {
        int i = e >> 6, k = e & 63;
        sX[i][k] = f2bf(0.0f);                 // zero K padding once
    }
    __syncthreads();

    // ---- hoist this wave's B-matrix fragments out of the time loop ----
    // B k order used by this code (same as A): lanes hold column N = tile*16 + lane%16;
    // lanes 0-15 carry K 0..15, lanes 16-31 carry K 16..31 of each 32-chunk.
    const int nrow = wv * 16 + (lane & 15);
    const int kb   = (lane >> 4) * 16;
    const v16bf bx0 = *(const v16bf*)&sWih[nrow][ 0 + kb];
    const v16bf bx1 = *(const v16bf*)&sWih[nrow][32 + kb];
    const v16bf bh  = *(const v16bf*)&sWhh[nrow][ 0 + kb];

    const int arow = lane & 15;
    const int ak   = (lane >> 4) * 16;
    const int dcol = wv * 16 + (lane & 15);
    const int mrow = (lane >> 4) * 8;

    for (int t = 0; t < Tc; ++t) {
        // ---- cooperative load of x[:, t, :] -> bf16 LDS tile ----
        // rows are 200B and 8B-aligned: float2 loads, packed b32 LDS stores.
        for (int e = tid; e < 16 * (Dc / 2); e += 192) {     // 400 float2
            int i = e / 25, k2 = e - i * 25;
            const float* p = xb + ((size_t)(soff + i) * Tc + t) * Dc + 2 * k2;
            float2 v = *(const float2*)p;
            *(unsigned*)&sX[i][2 * k2] = f2bf_pack(v.x, v.y);
            if (k2 == 0 && t + 1 < Tc) __builtin_prefetch(p + Dc, 0, 0); // next step
        }
        __syncthreads();

        // ---- A fragments (16x32 bf16 layout: lanes hold M, K split lo/hi) ----
        v16bf ax0 = cat8(*(const v8bf*)&sX[arow][ 0 + ak], *(const v8bf*)&sX[arow][ 8 + ak]);
        v16bf ax1 = cat8(*(const v8bf*)&sX[arow][32 + ak], *(const v8bf*)&sX[arow][40 + ak]);
        v16bf ah  = cat8(*(const v8bf*)&sH[arow][ 0 + ak], *(const v8bf*)&sH[arow][ 8 + ak]);

        v8f accX = {};
        v8f accH = {};
        accX = __builtin_amdgcn_wmma_f32_16x16x32_bf16(false, ax0, false, bx0, (short)0, accX, false, false);
        accX = __builtin_amdgcn_wmma_f32_16x16x32_bf16(false, ax1, false, bx1, (short)0, accX, false, false);
        accH = __builtin_amdgcn_wmma_f32_16x16x32_bf16(false, ah,  false, bh,  (short)0, accH, false, false);

#pragma unroll
        for (int v = 0; v < 8; ++v) {
            sSumX[mrow + v][dcol] = accX[v];
            sSumH[mrow + v][dcol] = accH[v];
        }
        __syncthreads();

        // ---- gates (fp32), torch gate order (r, z, n) ----
        for (int e = tid; e < 16 * Hc; e += 192) {
            int i = e / Hc, j = e - i * Hc;
            float xr = sSumX[i][j]      + sBih[j];
            float hr = sSumH[i][j]      + sBhh[j];
            float xz = sSumX[i][j + 30] + sBih[j + 30];
            float hz = sSumH[i][j + 30] + sBhh[j + 30];
            float xn = sSumX[i][j + 60] + sBih[j + 60];
            float hn = sSumH[i][j + 60] + sBhh[j + 60];
            float r = sigmoid_fast(xr + hr);
            float z = sigmoid_fast(xz + hz);
            float n = tanh_fast(xn + r * hn);
            float hv = (1.0f - z) * n + z * sHf[i][j];
            sHf[i][j] = hv;
            sH[i][j]  = f2bf(hv);
        }
        __syncthreads();
    }

    // ---- write final hidden: input1 half -> embeddings1 (d_out), input2 half -> ws ----
    for (int e = tid; e < 16 * Hc; e += 192) {
        int i = e / Hc, j = e - i * Hc;
        int n = s0 + i;
        float hv = sHf[i][j];
        if (n < HALFSEQ) emb1_out[(size_t)n * Hc + j] = hv;          // [b*300 + g*30 + j]
        else             emb2_ws[(size_t)(n - HALFSEQ) * Hc + j] = hv;
    }
}

// =====================================================================
// Kernel 2: tiny MLP head on embeddings2 (6 MFLOP total) -> logit.
// =====================================================================
__device__ __forceinline__ float gelu_exact(float x) {
    return 0.5f * x * (1.0f + erff(x * 0.70710678118654752f));
}

__global__ __launch_bounds__(32)
void mlp_kernel(const float* __restrict__ e2,
                const float* __restrict__ W1, const float* __restrict__ b1,
                const float* __restrict__ W2, const float* __restrict__ b2,
                const float* __restrict__ W3, const float* __restrict__ b3,
                float* __restrict__ logit)
{
    __shared__ float sE[Gc * Hc];   // 300
    __shared__ float sL1[EMB];
    __shared__ float sL2[NEMB];

    const int b = blockIdx.x, tid = threadIdx.x;
    for (int d = tid; d < Gc * Hc; d += 32) sE[d] = e2[(size_t)b * (Gc * Hc) + d];
    __syncthreads();

    if (tid < EMB) {
        float a = b1[tid];
        const float* w = W1 + (size_t)tid * (Gc * Hc);
        for (int k = 0; k < Gc * Hc; ++k) a += sE[k] * w[k];
        sL1[tid] = gelu_exact(a);
    }
    __syncthreads();

    if (tid < NEMB) {
        float a = b2[tid];
        const float* w = W2 + (size_t)tid * EMB;
        for (int k = 0; k < EMB; ++k) a += sL1[k] * w[k];
        sL2[tid] = gelu_exact(a);
    }
    __syncthreads();

    if (tid < OUT) {
        float a = b3[tid];
        const float* w = W3 + (size_t)tid * NEMB;
        for (int k = 0; k < NEMB; ++k) a += sL2[k] * w[k];
        logit[(size_t)b * OUT + tid] = a;
    }
}

// =====================================================================
extern "C" void kernel_launch(void* const* d_in, const int* in_sizes, int n_in,
                              void* d_out, int out_size, void* d_ws, size_t ws_size,
                              hipStream_t stream) {
    const float* in1 = (const float*)d_in[0];
    const float* in2 = (const float*)d_in[1];
    const float* Wih = (const float*)d_in[2];
    const float* Whh = (const float*)d_in[3];
    const float* bih = (const float*)d_in[4];
    const float* bhh = (const float*)d_in[5];
    const float* W1  = (const float*)d_in[6];
    const float* b1  = (const float*)d_in[7];
    const float* W2  = (const float*)d_in[8];
    const float* b2  = (const float*)d_in[9];
    const float* W3  = (const float*)d_in[10];
    const float* b3  = (const float*)d_in[11];

    float* out = (float*)d_out;               // [0,5120): logit, [5120,158720): embeddings1
    float* ws  = (float*)d_ws;                // embeddings2: 5120*30 floats = 614.4 KB

    gru_kernel<<<NGRP, 192, 0, stream>>>(in1, in2, Wih, Whh, bih, bhh,
                                         out + Bc * OUT, ws);
    mlp_kernel<<<Bc, 32, 0, stream>>>(ws, W1, b1, W2, b2, W3, b3, out);
}
